// DAGERC_fushion_73813307949180
// MI455X (gfx1250) — matrix operand order repaired and checked by reading, not compiled, not run on hardware
//
#include <hip/hip_runtime.h>
#include <hip/hip_bf16.h>

// ---------------- problem constants ----------------
#define HID    300
#define EMBD   1024
#define LAY    4
#define NCLS   7
#define BB     64
#define NN     64
#define G3     900                 // 3*HID
#define INDIM  2524                // HID*(LAY+1) + EMBD
#define ROWS   (BB*NN)             // 4096
#define NEGBIG 1e30f

typedef __attribute__((ext_vector_type(16))) _Float16 v16h;
typedef __attribute__((ext_vector_type(8)))  float    v8f;
typedef __attribute__((ext_vector_type(4)))  float    f32x4;

__device__ __forceinline__ float sigmoidf_(float x) { return 1.0f / (1.0f + __expf(-x)); }

// A fragment (16x32 f16): lane m=lane&15 is row; element e -> K = k0 + 16*hi + e
// -> one contiguous 16-float run at k0+16*hi (16B aligned), matching B.
__device__ __forceinline__ v16h load_a_frag(const float* __restrict__ Ar, int k0, int hi)
{
    const f32x4* p = (const f32x4*)(Ar + k0 + hi * 16);
    f32x4 x0 = p[0], x1 = p[1];    // k .. k+7
    f32x4 x2 = p[2], x3 = p[3];    // k+8 .. k+15
    v16h a;
    a[0]=(_Float16)x0.x; a[1]=(_Float16)x0.y; a[2]=(_Float16)x0.z; a[3]=(_Float16)x0.w;
    a[4]=(_Float16)x1.x; a[5]=(_Float16)x1.y; a[6]=(_Float16)x1.z; a[7]=(_Float16)x1.w;
    a[8]=(_Float16)x2.x; a[9]=(_Float16)x2.y; a[10]=(_Float16)x2.z; a[11]=(_Float16)x2.w;
    a[12]=(_Float16)x3.x; a[13]=(_Float16)x3.y; a[14]=(_Float16)x3.z; a[15]=(_Float16)x3.w;
    return a;
}

// B fragment (32x16 f16, B = W^T): lane n=lane&15 is column; element e -> K = k0+16*hi+e
// -> 16 contiguous halfs, 32B aligned (ldw % 16 == 0, k0 % 32 == 0).
__device__ __forceinline__ v16h load_b_frag(const _Float16* __restrict__ Wr, int k0, int hi)
{
    return *(const v16h*)(Wr + k0 + hi * 16);
}

// =====================================================================
// WMMA GEMM: C[M x Nc] = A[M x K](f32) * W(f16, padded Npad x Kpad)^T + bias
// 32x32 tile per wave (2x2 WMMA tiles), 8 waves/block. M % 32 == 0 required.
// Inner loop is branch-free: zero-padded weights neutralize A over-reads.
// =====================================================================
__global__ void __launch_bounds__(256)
wmma_gemm_kernel(const float* __restrict__ A, int lda,
                 const _Float16* __restrict__ W, int ldw,   // ldw = Kpad (mult of 32)
                 const float* __restrict__ bias,
                 float* __restrict__ C, int ldc,
                 int Mrows, int Ncols, int relu)
{
    const int lane = threadIdx.x & 31;
    const int wave = threadIdx.x >> 5;
    const int tilesN = (Ncols + 31) >> 5;          // 32-wide supertiles
    const int tilesM = Mrows >> 5;
    const int tile = blockIdx.x * 8 + wave;
    if (tile >= tilesM * tilesN) return;           // wave-uniform
    const int row0 = (tile / tilesN) << 5;
    const int col0 = (tile % tilesN) << 5;
    const int m  = lane & 15;
    const int hi = lane >> 4;

    const float*    Ar0 = A + (size_t)(row0 + m) * lda;
    const float*    Ar1 = A + (size_t)(row0 + 16 + m) * lda;
    const _Float16* Wr0 = W + (size_t)(col0 + m) * ldw;
    const _Float16* Wr1 = W + (size_t)(col0 + 16 + m) * ldw;

    v8f acc00 = {}, acc01 = {}, acc10 = {}, acc11 = {};
    for (int k0 = 0; k0 < ldw; k0 += 32) {
        v16h a0 = load_a_frag(Ar0, k0, hi);
        v16h a1 = load_a_frag(Ar1, k0, hi);
        v16h b0 = load_b_frag(Wr0, k0, hi);
        v16h b1 = load_b_frag(Wr1, k0, hi);
        acc00 = __builtin_amdgcn_wmma_f32_16x16x32_f16(false, a0, false, b0, (short)0, acc00, false, false);
        acc01 = __builtin_amdgcn_wmma_f32_16x16x32_f16(false, a0, false, b1, (short)0, acc01, false, false);
        acc10 = __builtin_amdgcn_wmma_f32_16x16x32_f16(false, a1, false, b0, (short)0, acc10, false, false);
        acc11 = __builtin_amdgcn_wmma_f32_16x16x32_f16(false, a1, false, b1, (short)0, acc11, false, false);
    }

    // C/D layout: VGPR rr -> row base + rr + 8*hi, column = cbase + (lane&15)
    const int cA = col0 + m, cB = col0 + 16 + m;
    const float bvA = (bias && cA < Ncols) ? bias[cA] : 0.0f;
    const float bvB = (bias && cB < Ncols) ? bias[cB] : 0.0f;
#pragma unroll
    for (int rr = 0; rr < 8; ++rr) {
        int rA = row0 + rr + 8 * hi;
        int rB = rA + 16;
        if (cA < Ncols) {
            float v0 = acc00[rr] + bvA; if (relu) v0 = fmaxf(v0, 0.0f);
            float v2 = acc10[rr] + bvA; if (relu) v2 = fmaxf(v2, 0.0f);
            C[(size_t)rA * ldc + cA] = v0;
            C[(size_t)rB * ldc + cA] = v2;
        }
        if (cB < Ncols) {
            float v1 = acc01[rr] + bvB; if (relu) v1 = fmaxf(v1, 0.0f);
            float v3 = acc11[rr] + bvB; if (relu) v3 = fmaxf(v3, 0.0f);
            C[(size_t)rA * ldc + cB] = v1;
            C[(size_t)rB * ldc + cB] = v3;
        }
    }
}

// ---------------- small helper kernels ----------------
// convert f32 weights [Nc x K] -> zero-padded f16 [Npad x Kpad]
__global__ void cvt_pad_kernel(const float* __restrict__ s, _Float16* __restrict__ d,
                               int Nc, int K, int Npad, int Kpad)
{
    int idx = blockIdx.x * blockDim.x + threadIdx.x;
    int total = Npad * Kpad;
    if (idx < total) {
        int n = idx / Kpad, k = idx - n * Kpad;
        d[idx] = (n < Nc && k < K) ? (_Float16)s[(size_t)n * K + k] : (_Float16)0.0f;
    }
}

__global__ void zero_kernel(float* __restrict__ p, int n)
{
    int i = blockIdx.x * blockDim.x + threadIdx.x;
    if (i < n) p[i] = 0.0f;
}

__global__ void copy_feat_kernel(const float* __restrict__ f, float* __restrict__ hcat)
{
    int idx = blockIdx.x * blockDim.x + threadIdx.x;
    if (idx < ROWS * EMBD) {
        int row = idx / EMBD, c = idx - row * EMBD;
        hcat[(size_t)row * INDIM + 5 * HID + c] = f[idx];
    }
}

__global__ void masks_kernel(const int* __restrict__ sp,
                             float* __restrict__ adj, float* __restrict__ smask)
{
    int b = blockIdx.x;
    int i = threadIdx.x;                       // 64 threads
    int si = sp[b * NN + i];
    bool vi = (si >= 0);
    int last = -1;
    for (int j = 0; j < i; ++j)
        if (sp[b * NN + j] == si) last = j;
    int start = last > 0 ? last : 0;
    for (int j = 0; j < NN; ++j) {
        int sj = sp[b * NN + j];
        bool vj = (sj >= 0);
        adj[((size_t)b * NN + i) * NN + j]   = (j < i && j >= start && vi) ? 1.0f : 0.0f;
        smask[((size_t)b * NN + i) * NN + j] = (si == sj && vi && vj) ? 1.0f : 0.0f;
    }
}

// qw[row] = Hin[row] . wq   (one wave per row)
__global__ void rowdot_kernel(const float* __restrict__ Hbase, const float* __restrict__ wq,
                              float* __restrict__ qw)
{
    int wid  = (blockIdx.x * blockDim.x + threadIdx.x) >> 5;
    int lane = threadIdx.x & 31;
    if (wid >= ROWS) return;
    const float* row = Hbase + (size_t)wid * INDIM;
    float s = 0.0f;
    for (int c = lane; c < HID; c += 32) s += row[c] * wq[c];
    for (int off = 16; off; off >>= 1) s += __shfl_xor(s, off, 32);
    if (lane == 0) qw[wid] = s;
}

// attention + message M for step i (one block per batch b)
__global__ void __launch_bounds__(256)
attn_m_kernel(const float* __restrict__ qw, const float* __restrict__ hk,
              const float* __restrict__ adj, const float* __restrict__ smask,
              const float* __restrict__ R0, const float* __restrict__ R1,
              const float* __restrict__ gatb, int l, int i,
              float* __restrict__ Mout)
{
    int b = blockIdx.x;
    int t = threadIdx.x;
    __shared__ float sh_w[NN];
    __shared__ float s_max, s_inv;
    if (t < NN) {
        float a  = adj[((size_t)b * NN + i) * NN + t];
        float al = qw[b * NN + i] + hk[b * NN + t] + gatb[l];
        al -= (1.0f - a) * NEGBIG;
        sh_w[t] = al;
    }
    __syncthreads();
    if (t == 0) {
        float mx = -INFINITY;
        for (int j = 0; j < NN; ++j) mx = fmaxf(mx, sh_w[j]);
        s_max = mx;
    }
    __syncthreads();
    if (t < NN) sh_w[t] = __expf(sh_w[t] - s_max);
    __syncthreads();
    if (t == 0) {
        float s = 0.0f;
        for (int j = 0; j < NN; ++j) s += sh_w[j];
        s_inv = 1.0f / s;
    }
    __syncthreads();
    for (int d = t; d < HID; d += blockDim.x) {
        float acc = 0.0f;
        for (int j = 0; j < i; ++j) {          // rows >= i carry zero weight
            float wj = sh_w[j] * s_inv;
            float sm = smask[((size_t)b * NN + i) * NN + j];
            float r0 = R0[((size_t)b * NN + j) * HID + d];
            float r1 = R1[((size_t)b * NN + j) * HID + d];
            acc += wj * (sm * r0 + (1.0f - sm) * r1);
        }
        Mout[b * HID + d] = acc;
    }
}

// GRU gates for both cells + write H1 row + incremental hk (one block per b)
__global__ void __launch_bounds__(256)
gates_kernel(const float* __restrict__ GIc, const float* __restrict__ GHp,
             const float* __restrict__ ghc, const float* __restrict__ gip,
             const float* __restrict__ bhh_c, const float* __restrict__ bih_p,
             const float* __restrict__ Mbuf, const float* __restrict__ wk,
             const float* __restrict__ Hin_base, float* __restrict__ Hout_base,
             float* __restrict__ hk, int i, int first)
{
    int b = blockIdx.x;
    const float* gi_c = GIc + ((size_t)b * NN + i) * G3;   // Q*wih_c^T + bih_c (precomputed)
    const float* gh_p = GHp + ((size_t)b * NN + i) * G3;   // Q*whh_p^T + bhh_p (precomputed)
    const float* q    = Hin_base + ((size_t)b * NN + i) * INDIM;
    float*       out  = Hout_base + ((size_t)b * NN + i) * INDIM;
    float partial = 0.0f;
    for (int c = threadIdx.x; c < HID; c += blockDim.x) {
        float hr, hz, hn, ir, iz, in_, mv;
        if (first) {
            hr = bhh_c[c]; hz = bhh_c[c + HID]; hn = bhh_c[c + 2 * HID];
            ir = bih_p[c]; iz = bih_p[c + HID]; in_ = bih_p[c + 2 * HID];
            mv = 0.0f;
        } else {
            const float* g = ghc + b * G3;     // M*whh_c^T + bhh_c
            const float* p = gip + b * G3;     // M*wih_p^T + bih_p
            hr = g[c]; hz = g[c + HID]; hn = g[c + 2 * HID];
            ir = p[c]; iz = p[c + HID]; in_ = p[c + 2 * HID];
            mv = Mbuf[b * HID + c];
        }
        float qv = q[c];
        // C = gru(x=Q, h=M)
        float r  = sigmoidf_(gi_c[c] + hr);
        float z  = sigmoidf_(gi_c[c + HID] + hz);
        float n  = tanhf(gi_c[c + 2 * HID] + r * hn);
        float Cv = (1.0f - z) * n + z * mv;
        // P = gru(x=M, h=Q)
        float rp = sigmoidf_(ir + gh_p[c]);
        float zp = sigmoidf_(iz + gh_p[c + HID]);
        float np = tanhf(in_ + rp * gh_p[c + 2 * HID]);
        float Pv = (1.0f - zp) * np + zp * qv;
        float o = Cv + Pv;
        out[c] = o;
        partial += o * wk[c];                  // maintain hk = H1 . wk incrementally
    }
    __shared__ float red[256];
    red[threadIdx.x] = partial;
    __syncthreads();
    for (int s = 128; s > 0; s >>= 1) {
        if (threadIdx.x < s) red[threadIdx.x] += red[threadIdx.x + s];
        __syncthreads();
    }
    if (threadIdx.x == 0) hk[b * NN + i] = red[0];
}

// =====================================================================
// host side
// =====================================================================
static inline int r32(int x) { return (x + 31) & ~31; }

static inline void launch_gemm(const float* A, int lda, const _Float16* W, int ldw,
                               const float* bias, float* C, int ldc,
                               int M, int Nc, int relu, hipStream_t s)
{
    int tiles  = (M >> 5) * (r32(Nc) >> 5);
    int blocks = (tiles + 7) / 8;
    wmma_gemm_kernel<<<blocks, 256, 0, s>>>(A, lda, W, ldw, bias, C, ldc, M, Nc, relu);
}

extern "C" void kernel_launch(void* const* d_in, const int* in_sizes, int n_in,
                              void* d_out, int out_size, void* d_ws, size_t ws_size,
                              hipStream_t stream)
{
    (void)in_sizes; (void)n_in; (void)out_size; (void)ws_size;
    const float* features = (const float*)d_in[0];
    const int*   speakers = (const int*)  d_in[1];
    const float* w_in  = (const float*)d_in[2];
    const float* b_in  = (const float*)d_in[3];
    const float* wq    = (const float*)d_in[4];
    const float* wk    = (const float*)d_in[5];
    const float* gatb  = (const float*)d_in[6];
    const float* wr0   = (const float*)d_in[7];
    const float* wr1   = (const float*)d_in[8];
    const float* wih_c = (const float*)d_in[9];
    const float* whh_c = (const float*)d_in[10];
    const float* bih_c = (const float*)d_in[11];
    const float* bhh_c = (const float*)d_in[12];
    const float* wih_p = (const float*)d_in[13];
    const float* whh_p = (const float*)d_in[14];
    const float* bih_p = (const float*)d_in[15];
    const float* bhh_p = (const float*)d_in[16];
    const float* w1 = (const float*)d_in[17];
    const float* b1 = (const float*)d_in[18];
    const float* w2 = (const float*)d_in[19];
    const float* b2 = (const float*)d_in[20];
    const float* w3 = (const float*)d_in[21];
    const float* b3 = (const float*)d_in[22];

    // padded weight geometry
    const int KP_H   = r32(HID);     // 320
    const int KP_E   = r32(EMBD);    // 1024
    const int KP_IN  = r32(INDIM);   // 2528
    const int NP_G3  = r32(G3);      // 928
    const int NP_H   = r32(HID);     // 320
    const int NP_C   = r32(NCLS);    // 32
    const size_t SZ_GATE = (size_t)NP_G3 * KP_H;   // per-layer padded wih/whh
    const size_t SZ_WR   = (size_t)NP_H  * KP_H;   // per-layer padded wr

    // ---- carve workspace ----
    char* wsp = (char*)d_ws;
    auto alloc = [&](size_t bytes) -> void* {
        void* p = wsp;
        wsp += (bytes + 255) & ~(size_t)255;
        return p;
    };
    float* hcat  = (float*)alloc((size_t)ROWS * INDIM * 4);
    float* gic   = (float*)alloc((size_t)ROWS * G3 * 4);      // reused as h1 after layers
    float* ghp   = (float*)alloc((size_t)ROWS * G3 * 4);      // reused as h2 after layers
    float* r0b   = (float*)alloc((size_t)ROWS * HID * 4);
    float* r1b   = (float*)alloc((size_t)ROWS * HID * 4);
    float* ghc   = (float*)alloc((size_t)BB * G3 * 4);
    float* gip   = (float*)alloc((size_t)BB * G3 * 4);
    float* mbuf  = (float*)alloc((size_t)BB * HID * 4);
    float* qwb   = (float*)alloc((size_t)ROWS * 4);
    float* hkb   = (float*)alloc((size_t)ROWS * 4);
    float* adj   = (float*)alloc((size_t)BB * NN * NN * 4);
    float* smask = (float*)alloc((size_t)BB * NN * NN * 4);
    _Float16* w_in_h  = (_Float16*)alloc((size_t)NP_H * KP_E * 2);
    _Float16* wih_c_h = (_Float16*)alloc(LAY * SZ_GATE * 2);
    _Float16* whh_c_h = (_Float16*)alloc(LAY * SZ_GATE * 2);
    _Float16* wih_p_h = (_Float16*)alloc(LAY * SZ_GATE * 2);
    _Float16* whh_p_h = (_Float16*)alloc(LAY * SZ_GATE * 2);
    _Float16* wr0_h   = (_Float16*)alloc(LAY * SZ_WR * 2);
    _Float16* wr1_h   = (_Float16*)alloc(LAY * SZ_WR * 2);
    _Float16* w1_h    = (_Float16*)alloc((size_t)NP_H * KP_IN * 2);
    _Float16* w2_h    = (_Float16*)alloc((size_t)NP_H * KP_H * 2);
    _Float16* w3_h    = (_Float16*)alloc((size_t)NP_C * KP_H * 2);

    auto cvt = [&](const float* s, _Float16* d, int Nc, int K, int Npad, int Kpad) {
        int n = Npad * Kpad;
        cvt_pad_kernel<<<(n + 255) / 256, 256, 0, stream>>>(s, d, Nc, K, Npad, Kpad);
    };

    // ---- one-time prep (deterministic; re-run every call) ----
    masks_kernel<<<BB, NN, 0, stream>>>(speakers, adj, smask);
    copy_feat_kernel<<<(ROWS * EMBD + 255) / 256, 256, 0, stream>>>(features, hcat);
    cvt(w_in, w_in_h, HID, EMBD, NP_H, KP_E);
    for (int l = 0; l < LAY; ++l) {
        cvt(wih_c + (size_t)l * G3 * HID, wih_c_h + l * SZ_GATE, G3, HID, NP_G3, KP_H);
        cvt(whh_c + (size_t)l * G3 * HID, whh_c_h + l * SZ_GATE, G3, HID, NP_G3, KP_H);
        cvt(wih_p + (size_t)l * G3 * HID, wih_p_h + l * SZ_GATE, G3, HID, NP_G3, KP_H);
        cvt(whh_p + (size_t)l * G3 * HID, whh_p_h + l * SZ_GATE, G3, HID, NP_G3, KP_H);
        cvt(wr0 + (size_t)l * HID * HID, wr0_h + l * SZ_WR, HID, HID, NP_H, KP_H);
        cvt(wr1 + (size_t)l * HID * HID, wr1_h + l * SZ_WR, HID, HID, NP_H, KP_H);
    }
    cvt(w1, w1_h, HID, INDIM, NP_H, KP_IN);
    cvt(w2, w2_h, HID, HID, NP_H, KP_H);
    cvt(w3, w3_h, NCLS, HID, NP_C, KP_H);

    // ---- H0 = relu(features @ w_in^T + b_in) -> Hcat segment 0 ----
    launch_gemm(features, EMBD, w_in_h, KP_E, b_in, hcat, INDIM, ROWS, HID, 1, stream);

    // ---- GNN layers ----
    for (int l = 0; l < LAY; ++l) {
        const float* Hin  = hcat + (size_t)HID * l;
        float*       Hout = hcat + (size_t)HID * (l + 1);
        const _Float16* Wihc = wih_c_h + l * SZ_GATE;
        const _Float16* Whhc = whh_c_h + l * SZ_GATE;
        const _Float16* Wihp = wih_p_h + l * SZ_GATE;
        const _Float16* Whhp = whh_p_h + l * SZ_GATE;
        const _Float16* Wr0  = wr0_h + l * SZ_WR;
        const _Float16* Wr1  = wr1_h + l * SZ_WR;

        // hoisted Q-dependent GRU matmuls (big parallel GEMMs)
        launch_gemm(Hin, INDIM, Wihc, KP_H, bih_c + l * G3, gic, G3, ROWS, G3, 0, stream);
        launch_gemm(Hin, INDIM, Whhp, KP_H, bhh_p + l * G3, ghp, G3, ROWS, G3, 0, stream);
        rowdot_kernel<<<(ROWS * 32 + 255) / 256, 256, 0, stream>>>(Hin, wq + l * HID, qwb);
        zero_kernel<<<(ROWS + 255) / 256, 256, 0, stream>>>(hkb, ROWS);

        // step 0: C0 = gru(x0, 0), P0 = gru(0, x0)
        gates_kernel<<<BB, 256, 0, stream>>>(gic, ghp, ghc, gip,
                                             bhh_c + l * G3, bih_p + l * G3,
                                             mbuf, wk + l * HID, Hin, Hout, hkb, 0, 1);
        launch_gemm(Hout, NN * INDIM, Wr0, KP_H, nullptr, r0b, NN * HID, BB, HID, 0, stream);
        launch_gemm(Hout, NN * INDIM, Wr1, KP_H, nullptr, r1b, NN * HID, BB, HID, 0, stream);

        for (int i = 1; i < NN; ++i) {
            attn_m_kernel<<<BB, 256, 0, stream>>>(qwb, hkb, adj, smask, r0b, r1b,
                                                  gatb, l, i, mbuf);
            // M-dependent GRU gate GEMMs (64 x 900, K = 300)
            launch_gemm(mbuf, HID, Whhc, KP_H, bhh_c + l * G3, ghc, G3, BB, G3, 0, stream);
            launch_gemm(mbuf, HID, Wihp, KP_H, bih_p + l * G3, gip, G3, BB, G3, 0, stream);
            gates_kernel<<<BB, 256, 0, stream>>>(gic, ghp, ghc, gip, nullptr, nullptr,
                                                 mbuf, wk + l * HID, Hin, Hout, hkb, i, 0);
            // incremental R0/R1 for new row i (64 x 300, K = 300)
            launch_gemm(Hout + (size_t)i * INDIM, NN * INDIM, Wr0, KP_H, nullptr,
                        r0b + (size_t)i * HID, NN * HID, BB, HID, 0, stream);
            launch_gemm(Hout + (size_t)i * INDIM, NN * INDIM, Wr1, KP_H, nullptr,
                        r1b + (size_t)i * HID, NN * HID, BB, HID, 0, stream);
        }
    }

    // ---- final MLP over Hcat (reuse gic/ghp buffers) ----
    float* h1 = gic;
    float* h2 = ghp;
    launch_gemm(hcat, INDIM, w1_h, KP_IN, b1, h1, HID, ROWS, HID, 1, stream);
    launch_gemm(h1, HID, w2_h, KP_H, b2, h2, HID, ROWS, HID, 1, stream);
    launch_gemm(h2, HID, w3_h, KP_H, b3, (float*)d_out, NCLS, ROWS, NCLS, 0, stream);
}
